// MultiHeadAttention_9062380995269
// MI455X (gfx1250) — matrix operand order repaired and checked by reading, not compiled, not run on hardware
//
#include <hip/hip_runtime.h>

// MHA with band mask for MI455X (gfx1250, wave32, WMMA + TDM).
// B=2, S=2048, D=1024, H=16, DK=64, band half-width = 64.
// d_out = out [B,S,D] fp32 ++ attn [B,H,S,S] fp32 (zero outside band).
// Output-bandwidth bound (536MB attn write). GEMMs: v_wmma_f32_16x16x32_bf16,
// tiles staged LDS<-global via tensor_load_to_lds (Tensor Data Mover).

#define B_    2
#define S_    2048
#define D_    1024
#define H_    16
#define DK_   64
#define HALF_ 64
#define NEG_  (-1e9f)

typedef __attribute__((ext_vector_type(16))) __bf16          v16bf;
typedef __attribute__((ext_vector_type(16))) unsigned short  v16u;
typedef __attribute__((ext_vector_type(8)))  unsigned short  v8us;
typedef __attribute__((ext_vector_type(4)))  unsigned short  v4us;
typedef __attribute__((ext_vector_type(8)))  float           v8f;
typedef __attribute__((ext_vector_type(4)))  unsigned int    tdm_g0_t;
typedef __attribute__((ext_vector_type(8)))  int             tdm_g1_t;

static __device__ __forceinline__ unsigned short f2bf(float f) {
    union { float f; unsigned u; } x; x.f = f;
    unsigned r = x.u + 0x7FFFu + ((x.u >> 16) & 1u);   // round-to-nearest-even
    return (unsigned short)(r >> 16);
}

static __device__ __forceinline__ v8f v8f_zero() {
    v8f z = {0.f, 0.f, 0.f, 0.f, 0.f, 0.f, 0.f, 0.f};
    return z;
}

static __device__ __forceinline__ v16u mk16(v8us lo, v8us hi) {
    union { struct { v8us a, b; } s; v16u v; } u;
    u.s.a = lo; u.s.b = hi;
    return u.v;
}

static __device__ __forceinline__ v8f wmma_bf16(const v16u& a, const v16u& b, const v8f& c) {
    union { v16u u; v16bf h; } ua, ub;
    ua.u = a; ub.u = b;
    return __builtin_amdgcn_wmma_f32_16x16x32_bf16(false, ua.h, false, ub.h,
                                                   (short)0, c, false, false);
}

static __device__ __forceinline__ unsigned lds_off(const void* p) {
    // addrspace(3) offset == low 32 bits of the generic pointer
    return (unsigned)(unsigned long long)p;
}

// ---------------------------------------------------------------------------
// Tensor Data Mover: DMA a 2D bf16 tile (tile_d1 rows x tile_d0 elems) from
// global into LDS. pad_enable inserts 4 DWORDs (16B) after every 16 DWORDs
// (64B = one 32-elem bf16 row) -> padded LDS row stride of 80B, matching the
// [rows][40] ushort LDS tiles used by the WMMA fragment loads.
// D# layout per CDNA5 ISA. Issued per-wave
// (TDM ignores EXEC); pair with s_wait_tensorcnt + barrier.
// ---------------------------------------------------------------------------
static __device__ __forceinline__ void tdm_load_2d_bf16(
    unsigned ldsByteOff, const void* gptr,
    unsigned tensor_d0, unsigned tensor_d1,
    unsigned tile_d0, unsigned tile_d1, unsigned row_stride_elems)
{
    unsigned long long ga = (unsigned long long)gptr;

    tdm_g0_t g0;
    g0[0] = 1u;                                        // count=1, user descriptor
    g0[1] = ldsByteOff;                                // lds_addr
    g0[2] = (unsigned)(ga & 0xFFFFFFFFu);              // global_addr[31:0]
    g0[3] = (unsigned)((ga >> 32) & 0x1FFFFFFu)        // global_addr[56:32]
          | 0x80000000u;                               // type=2 ("image")

    tdm_g1_t g1;
    g1[0] = (int)((1u << 16)                           // data_size = 2 bytes
                | (1u << 20)                           // pad_enable
                | (3u << 22)                           // pad_interval: 16 DWORDs
                | (3u << 25));                         // pad_amount: 4 DWORDs
    g1[1] = (int)((tensor_d0 & 0xFFFFu) << 16);        // tensor_dim0[15:0]
    g1[2] = (int)((tensor_d0 >> 16) | ((tensor_d1 & 0xFFFFu) << 16));
    g1[3] = (int)((tensor_d1 >> 16) | (tile_d0 << 16));// tile_dim0
    g1[4] = (int)(tile_d1 & 0xFFFFu);                  // tile_dim1 (tile_dim2=0)
    g1[5] = (int)row_stride_elems;                     // tensor_dim0_stride[31:0]
    g1[6] = 0;
    g1[7] = 0;

    asm volatile("tensor_load_to_lds %0, %1" :: "s"(g0), "s"(g1) : "memory");
}

// ---------------------------------------------------------------------------
// Zero fill for the 536MB attn region (band entries overwritten later).
// ---------------------------------------------------------------------------
__global__ void zero_f4_kernel(float4* __restrict__ p, size_t n4) {
    size_t i      = (size_t)blockIdx.x * blockDim.x + threadIdx.x;
    size_t stride = (size_t)gridDim.x * blockDim.x;
    const float4 z = make_float4(0.f, 0.f, 0.f, 0.f);
    for (; i < n4; i += stride) p[i] = z;
}

// ---------------------------------------------------------------------------
// Elementwise fp32 -> bf16 (activations). n4 = element count / 4.
// ---------------------------------------------------------------------------
__global__ void cvt_bf16_kernel(const float* __restrict__ x,
                                unsigned short* __restrict__ y, size_t n4) {
    size_t i      = (size_t)blockIdx.x * blockDim.x + threadIdx.x;
    size_t stride = (size_t)gridDim.x * blockDim.x;
    for (; i < n4; i += stride) {
        float4 v = ((const float4*)x)[i];
        v4us o = { f2bf(v.x), f2bf(v.y), f2bf(v.z), f2bf(v.w) };
        ((v4us*)y)[i] = o;
    }
}

// ---------------------------------------------------------------------------
// Weight transpose + convert: W [K=D][N=D] fp32 -> Wt [N][K] bf16.
// ---------------------------------------------------------------------------
__global__ __launch_bounds__(256) void transpose_w_kernel(
    const float* __restrict__ W, unsigned short* __restrict__ Wt)
{
    __shared__ float tile[32][33];
    const int tx = threadIdx.x & 31;
    const int ty = threadIdx.x >> 5;             // 0..7
    const int n0 = blockIdx.x * 32;
    const int k0 = blockIdx.y * 32;
    #pragma unroll
    for (int j = 0; j < 4; ++j)
        tile[ty + j * 8][tx] = W[(size_t)(k0 + ty + j * 8) * D_ + n0 + tx];
    __syncthreads();
    #pragma unroll
    for (int j = 0; j < 4; ++j)
        Wt[(size_t)(n0 + ty + j * 8) * D_ + k0 + tx] = f2bf(tile[tx][ty + j * 8]);
}

// ---------------------------------------------------------------------------
// GEMM: C[M,N] = A[M,K] @ Wt^T + bias, A bf16 [M][K], Wt bf16 [N][K].
// M=4096, N=K=1024. Block 256 = 8 waves; tile 128x128x32; wave = 32x64.
// Tiles staged via TDM: wave w DMAs its 16-row slice of As and Bs.
// HEADED=1 -> bf16 out scattered to [B,H,S,DK]; HEADED=0 -> fp32 [M,N].
// ---------------------------------------------------------------------------
template <int HEADED>
__global__ __launch_bounds__(256) void gemm_tdm_kernel(
    const unsigned short* __restrict__ A, const unsigned short* __restrict__ Wt,
    const float* __restrict__ bias, void* __restrict__ Cout,
    int M, int N, int K)
{
    __shared__ unsigned short As[128][40];   // [m][k], 80B padded rows (TDM pad)
    __shared__ unsigned short Bs[128][40];   // [n][k]

    const int t    = threadIdx.x;
    const int m0   = blockIdx.y * 128;
    const int n0   = blockIdx.x * 128;
    const int wv   = t >> 5;
    const int lane = t & 31;
    const int lm   = lane & 15;
    const int hi   = lane >> 4;
    const int c0   = hi * 8;
    const int wm   = (wv >> 1) * 32;
    const int wn   = (wv & 1) * 64;

    const unsigned asSlice = lds_off(&As[wv * 16][0]);
    const unsigned bsSlice = lds_off(&Bs[wv * 16][0]);
    const unsigned short* aRow = A  + (size_t)(m0 + wv * 16) * K;
    const unsigned short* bRow = Wt + (size_t)(n0 + wv * 16) * K;

    v8f acc[2][4];
    #pragma unroll
    for (int i = 0; i < 2; ++i)
        #pragma unroll
        for (int j = 0; j < 4; ++j) acc[i][j] = v8f_zero();

    for (int k0 = 0; k0 < K; k0 += 32) {
        __syncthreads();                         // LDS free for rewrite
        tdm_load_2d_bf16(asSlice, aRow + k0, K, 16, 32, 16, K);
        tdm_load_2d_bf16(bsSlice, bRow + k0, K, 16, 32, 16, K);
        __builtin_amdgcn_s_wait_tensorcnt(0);
        __syncthreads();                         // all 8 slices visible

        v16u af[2];
        #pragma unroll
        for (int ms = 0; ms < 2; ++ms) {
            const unsigned short* p = &As[wm + ms * 16 + lm][hi * 16];
            #pragma unroll
            for (int i = 0; i < 8; ++i) { af[ms][i] = p[i]; af[ms][8 + i] = p[8 + i]; }
        }
        v16u bfr[4];
        #pragma unroll
        for (int ns = 0; ns < 4; ++ns) {
            const unsigned short* p = &Bs[wn + ns * 16 + lm][hi * 16];
            #pragma unroll
            for (int i = 0; i < 16; ++i) bfr[ns][i] = p[i];
        }
        #pragma unroll
        for (int ms = 0; ms < 2; ++ms)
            #pragma unroll
            for (int ns = 0; ns < 4; ++ns)
                acc[ms][ns] = wmma_bf16(af[ms], bfr[ns], acc[ms][ns]);
    }

    // Epilogue with precomputed bases (no per-element div/mod).
    if (HEADED) {
        const int bb  = m0 >> 11;                 // S_=2048 rows per batch
        const int hh  = (n0 + wn) >> 6;           // DK_=64 cols per head
        const int ssb = (m0 & (S_ - 1)) + wm + hi * 8;
        unsigned short* O = (unsigned short*)Cout;
        const size_t headBase = (size_t)(bb * H_ + hh) * S_ * DK_;
        #pragma unroll
        for (int ms = 0; ms < 2; ++ms)
            #pragma unroll
            for (int ns = 0; ns < 4; ++ns) {
                const float bv = bias[n0 + wn + ns * 16 + lm];
                #pragma unroll
                for (int r = 0; r < 8; ++r) {
                    size_t idx = headBase + (size_t)(ssb + ms * 16 + r) * DK_ + ns * 16 + lm;
                    O[idx] = f2bf(acc[ms][ns][r] + bv);
                }
            }
    } else {
        float* O = (float*)Cout;
        #pragma unroll
        for (int ms = 0; ms < 2; ++ms)
            #pragma unroll
            for (int ns = 0; ns < 4; ++ns) {
                const int   n  = n0 + wn + ns * 16 + lm;
                const float bv = bias[n];
                #pragma unroll
                for (int r = 0; r < 8; ++r) {
                    const int m = m0 + wm + ms * 16 + r + hi * 8;
                    O[(size_t)m * N + n] = acc[ms][ns][r] + bv;
                }
            }
    }
}

// ---------------------------------------------------------------------------
// Banded attention. Grid (S/64, H, B), block 128 = 4 waves, 1 q-tile/wave.
// Qp/Kp/Vp are bf16 [B,H,S,DK]. Writes fp32 probs to attn band and bf16
// context rows to Xc [B*S, D].
// ---------------------------------------------------------------------------
__global__ __launch_bounds__(128) void band_attention_kernel(
    const unsigned short* __restrict__ Qp, const unsigned short* __restrict__ Kp,
    const unsigned short* __restrict__ Vp, float* __restrict__ attnOut,
    unsigned short* __restrict__ Xc)
{
    __shared__ unsigned short probs[4][16][168];  // per wave: 16 x 160 keys (+pad)

    const int wv   = threadIdx.x >> 5;
    const int lane = threadIdx.x & 31;
    const int lm   = lane & 15;
    const int hi   = lane >> 4;
    const int c0   = hi * 8;

    const int b  = blockIdx.z;
    const int h  = blockIdx.y;
    const int q0 = (blockIdx.x * 4 + wv) * 16;

    const size_t head = (size_t)(b * H_ + h) * S_ * DK_;
    const unsigned short* Qb = Qp + head;
    const unsigned short* Kb = Kp + head;
    const unsigned short* Vb = Vp + head;

    // Q fragments (A layout): row lm, K = hi*16 .. hi*16+15 contiguous
    v16u aq[2];
    #pragma unroll
    for (int kc = 0; kc < 2; ++kc) {
        const unsigned short* qr = Qb + (size_t)(q0 + lm) * DK_ + kc * 32;
        aq[kc] = mk16(*(const v8us*)(qr + hi * 16), *(const v8us*)(qr + hi * 16 + 8));
    }

    // Scores: 9 key tiles, band-masked
    float sc[9][8];
    #pragma unroll
    for (int tt = 0; tt < 9; ++tt) {
        const int jb = q0 - HALF_ + tt * 16;
        int j  = jb + lm;
        int jc = j < 0 ? 0 : (j >= S_ ? S_ - 1 : j);
        v16u bk[2];
        #pragma unroll
        for (int kc = 0; kc < 2; ++kc) {
            const unsigned short* kr = Kb + (size_t)jc * DK_ + kc * 32 + hi * 16;
            bk[kc] = mk16(*(const v8us*)kr, *(const v8us*)(kr + 8));
        }
        v8f cacc = v8f_zero();
        cacc = wmma_bf16(aq[0], bk[0], cacc);
        cacc = wmma_bf16(aq[1], bk[1], cacc);
        #pragma unroll
        for (int r = 0; r < 8; ++r) {
            int i = q0 + r + hi * 8;
            int jcol = jb + lm;
            float s = cacc[r] * 0.125f;            // 1/sqrt(64)
            bool ok = (jcol >= 0) && (jcol < S_) &&
                      (i - jcol <= HALF_) && (jcol - i <= HALF_);
            sc[tt][r] = ok ? s : NEG_;
        }
    }

    // Softmax per row (16-lane butterfly)
    float inv[8];
    #pragma unroll
    for (int r = 0; r < 8; ++r) {
        float mx = sc[0][r];
        #pragma unroll
        for (int tt = 1; tt < 9; ++tt) mx = fmaxf(mx, sc[tt][r]);
        #pragma unroll
        for (int d = 1; d < 16; d <<= 1) mx = fmaxf(mx, __shfl_xor(mx, d, 32));
        float sum = 0.f;
        #pragma unroll
        for (int tt = 0; tt < 9; ++tt) {
            float p = __expf(sc[tt][r] - mx);
            sc[tt][r] = p;
            sum += p;
        }
        #pragma unroll
        for (int d = 1; d < 16; d <<= 1) sum += __shfl_xor(sum, d, 32);
        inv[r] = 1.0f / sum;
    }

    // Emit fp32 probs to attn band; stage bf16 probs in LDS
    const size_t arow = (size_t)(b * H_ + h) * S_;
    #pragma unroll
    for (int tt = 0; tt < 9; ++tt) {
        const int jb   = q0 - HALF_ + tt * 16;
        const int jcol = jb + lm;
        #pragma unroll
        for (int r = 0; r < 8; ++r) {
            const int   irow = r + hi * 8;
            const float pr   = sc[tt][r] * inv[r];
            if (jcol >= 0 && jcol < S_)
                attnOut[(arow + q0 + irow) * (size_t)S_ + jcol] = pr;
            probs[wv][irow][tt * 16 + lm] = f2bf(pr);
        }
    }
    #pragma unroll
    for (int r = 0; r < 8; ++r) probs[wv][r + hi * 8][144 + lm] = 0;
    __syncthreads();

    // attn @ V: K = 5 chunks of 32 keys, N = 4 tiles of 16
    #pragma unroll
    for (int nt = 0; nt < 4; ++nt) {
        v8f acc = v8f_zero();
        #pragma unroll
        for (int kc = 0; kc < 5; ++kc) {
            const unsigned short* pp = &probs[wv][lm][kc * 32 + hi * 16];
            v16u pa = mk16(*(const v8us*)pp, *(const v8us*)(pp + 8));
            v16u vbv;
            const int n = nt * 16 + lm;
            #pragma unroll
            for (int i = 0; i < 16; ++i) {
                int kk = kc * 32 + hi * 16 + i;
                int j  = q0 - HALF_ + kk;
                j = j < 0 ? 0 : (j >= S_ ? S_ - 1 : j);   // padded probs are 0
                vbv[i] = Vb[(size_t)j * DK_ + n];
            }
            acc = wmma_bf16(pa, vbv, acc);
        }
        #pragma unroll
        for (int r = 0; r < 8; ++r) {
            int i   = q0 + r + hi * 8;
            int col = h * DK_ + nt * 16 + lm;
            Xc[(size_t)(b * S_ + i) * D_ + col] = f2bf(acc[r]);
        }
    }
}

// ---------------------------------------------------------------------------
extern "C" void kernel_launch(void* const* d_in, const int* in_sizes, int n_in,
                              void* d_out, int out_size, void* d_ws, size_t ws_size,
                              hipStream_t stream) {
    const float* query = (const float*)d_in[0];
    const float* key   = (const float*)d_in[1];
    const float* value = (const float*)d_in[2];
    const float* Wq    = (const float*)d_in[3];
    const float* bq    = (const float*)d_in[4];
    const float* Wk    = (const float*)d_in[5];
    const float* bk    = (const float*)d_in[6];
    const float* Wv    = (const float*)d_in[7];
    const float* bv    = (const float*)d_in[8];
    const float* Wo    = (const float*)d_in[9];
    const float* bo    = (const float*)d_in[10];

    float* out  = (float*)d_out;
    float* attn = out + (size_t)B_ * S_ * D_;

    // Workspace (all bf16/ushort): qbf|kbf|vbf|Qp|Kp|Vp|Xc (4M elems each)
    // + Wtq|Wtk|Wtv|Wto (1M elems each) = 64MB total.
    const size_t per = (size_t)B_ * S_ * D_;   // 4,194,304
    unsigned short* ws  = (unsigned short*)d_ws;
    unsigned short* qbf = ws;
    unsigned short* kbf = qbf + per;
    unsigned short* vbf = kbf + per;
    unsigned short* Qp  = vbf + per;
    unsigned short* Kp  = Qp + per;
    unsigned short* Vp  = Kp + per;
    unsigned short* Xc  = Vp + per;
    unsigned short* Wtq = Xc + per;
    unsigned short* Wtk = Wtq + (size_t)D_ * D_;
    unsigned short* Wtv = Wtk + (size_t)D_ * D_;
    unsigned short* Wto = Wtv + (size_t)D_ * D_;

    const size_t attnN4 = (size_t)B_ * H_ * S_ * S_ / 4;
    zero_f4_kernel<<<8192, 256, 0, stream>>>((float4*)attn, attnN4);

    cvt_bf16_kernel<<<2048, 256, 0, stream>>>(query, qbf, per / 4);
    cvt_bf16_kernel<<<2048, 256, 0, stream>>>(key,   kbf, per / 4);
    cvt_bf16_kernel<<<2048, 256, 0, stream>>>(value, vbf, per / 4);

    dim3 tg(D_ / 32, D_ / 32);
    transpose_w_kernel<<<tg, 256, 0, stream>>>(Wq, Wtq);
    transpose_w_kernel<<<tg, 256, 0, stream>>>(Wk, Wtk);
    transpose_w_kernel<<<tg, 256, 0, stream>>>(Wv, Wtv);
    transpose_w_kernel<<<tg, 256, 0, stream>>>(Wo, Wto);

    dim3 gg(D_ / 128, (B_ * S_) / 128);   // (8, 32)
    gemm_tdm_kernel<1><<<gg, 256, 0, stream>>>(qbf, Wtq, bq, Qp, B_ * S_, D_, D_);
    gemm_tdm_kernel<1><<<gg, 256, 0, stream>>>(kbf, Wtk, bk, Kp, B_ * S_, D_, D_);
    gemm_tdm_kernel<1><<<gg, 256, 0, stream>>>(vbf, Wtv, bv, Vp, B_ * S_, D_, D_);

    band_attention_kernel<<<dim3(S_ / 64, H_, B_), 128, 0, stream>>>(Qp, Kp, Vp, attn, Xc);

    gemm_tdm_kernel<0><<<gg, 256, 0, stream>>>(Xc, Wto, bo, out, B_ * S_, D_, D_);
}
